// SVGraphConvLayer_17892833755726
// MI455X (gfx1250) — matrix operand order repaired and checked by reading, not compiled, not run on hardware
//
#include <hip/hip_runtime.h>
#include <math.h>

#define TPB 128
#define WAVES 4

typedef __attribute__((ext_vector_type(16))) __bf16 v16bf;
typedef __attribute__((ext_vector_type(8))) float v8f;

// ---------- bf16 helpers (raw-bit storage as unsigned short) ----------
__device__ __forceinline__ unsigned short f2bf(float f) {
  unsigned int u = __float_as_uint(f);
  unsigned int r = u + 0x7FFFu + ((u >> 16) & 1u);
  return (unsigned short)(r >> 16);
}
__device__ __forceinline__ float bf2f(unsigned short s) {
  return __uint_as_float(((unsigned int)s) << 16);
}
__device__ __forceinline__ __bf16 bfbits(unsigned short s) {
  union { unsigned short u; __bf16 b; } x; x.u = s; return x.b;
}
__device__ __forceinline__ void atomicMaxF(float* addr, float v) {
  if (v >= 0.f) atomicMax((int*)addr, __float_as_int(v));
  else          atomicMin((unsigned int*)addr, __float_as_uint(v));
}

// ---------- WMMA tile: C(16x16,f32) = A(16xK from LDS) * B(Kx16), B col n = weight row n ----------
// A layout (16-bit A 16x32): lane<16 row=l16 holds K {k0..k0+7, k0+16..k0+23};
// lane>=16 holds K {k0+8..k0+15, k0+24..k0+31}.
// B layout: lane l holds column N=l%16; same K order as A (half*8 offset, +16 for upper 8).
__device__ __forceinline__ v8f wmma_tile(const unsigned short* A, int ldA, int arow,
                                         const unsigned short* W, int ldB, int nbase,
                                         int kpad, int lane) {
  v8f acc = {0.f, 0.f, 0.f, 0.f, 0.f, 0.f, 0.f, 0.f};
  const int half = (lane >> 4) & 1;
  const int l16 = lane & 15;
  const unsigned short* ap = A + (size_t)(arow + l16) * ldA + half * 8;
  const unsigned short* bp = W + (size_t)(nbase + l16) * ldB + half * 8;
  for (int k0 = 0; k0 < kpad; k0 += 32) {
    v16bf av, bv;
#pragma unroll
    for (int t = 0; t < 8; ++t) av[t] = bfbits(ap[k0 + t]);
#pragma unroll
    for (int t = 0; t < 8; ++t) av[8 + t] = bfbits(ap[k0 + 16 + t]);
#pragma unroll
    for (int t = 0; t < 8; ++t) { bv[t] = bfbits(bp[k0 + t]); bv[8 + t] = bfbits(bp[k0 + 16 + t]); }
    acc = __builtin_amdgcn_wmma_f32_16x16x32_bf16(false, av, false, bv,
                                                  (short)0, acc, false, false);
  }
  return acc;
}

// C element (VGPR r, lane): row = half*8+r, col = lane%16
template <typename EPI>
__device__ __forceinline__ void gemm_grid(const unsigned short* A, int ldA, int mtiles,
                                          const unsigned short* W, int ldB, int ntiles,
                                          int kpad, int wave, int lane, EPI epi) {
  const int half = (lane >> 4) & 1;
  const int l16 = lane & 15;
  for (int t = wave; t < mtiles * ntiles; t += WAVES) {
    const int mt = t / ntiles, nt = t - mt * ntiles;
    v8f c = wmma_tile(A, ldA, mt * 16, W, ldB, nt * 16, kpad, lane);
#pragma unroll
    for (int r = 0; r < 8; ++r) epi(mt * 16 + half * 8 + r, nt * 16 + l16, c[r]);
  }
}

// ---------- packed weights per SVP ----------
struct SVPW {
  const unsigned short *Wh, *Wu, *Ws, *Wg;
  const float *bs, *bg;
  int kv;    // padded in_v (K of Wh gemm, ldB of Wh)
  int h;     // 16 or 48
  int hpad;  // 32 or 64 (K of Wu)
  int ins;   // scalar input width
  int kspad; // padded ins+h = stride of Ascat and ldB of Ws
};
struct WAll {
  SVPW tgt, src, edge, out1, out2;
  const unsigned short *v2s, *s2v;
};

// ---------- one SVP on a 16-edge tile (block-cooperative) ----------
// Avec: 48x64 bf16, rows (c*16+n), cols [0,in_v) = local-frame vectors, rest zero
// Ascat: 16 x kspad bf16, cols [0,ins) filled; fn fills nrm at [ins, ins+h); pad zero
// vh: f32 scratch [48][48]; gate: f32 [16][16]
// sout_f (optional f32 16x128), sout_b (bf16, stride ldSB), vout f32 [(c*16+n)*16+o]
__device__ void run_svp(const SVPW& w,
                        unsigned short* Avec, int ldV,
                        unsigned short* Ascat,
                        float* vh, float* gate,
                        float* sout_f, unsigned short* sout_b, int ldSB,
                        float* vout, int wave, int lane, int tid) {
  const int nth = w.h >> 4;
  // vh = Wh * v (rows = (c,n), N = h)
  gemm_grid(Avec, ldV, 3, w.Wh, w.kv, nth, w.kv, wave, lane,
            [&](int row, int col, float val) { vh[row * 48 + col] = val; });
  __syncthreads();
  // norms -> Ascat[:, ins..ins+h) ; vh bf16 -> Avec (pads beyond h stay zero)
  for (int idx = tid; idx < 16 * w.h; idx += TPB) {
    int n = idx / w.h, hc = idx - n * w.h;
    float a0 = vh[n * 48 + hc];
    float a1 = vh[(16 + n) * 48 + hc];
    float a2 = vh[(32 + n) * 48 + hc];
    Ascat[n * w.kspad + w.ins + hc] = f2bf(sqrtf(a0 * a0 + a1 * a1 + a2 * a2 + 1e-8f));
  }
  for (int idx = tid; idx < 48 * w.h; idx += TPB) {
    int row = idx / w.h, hc = idx - row * w.h;
    Avec[row * ldV + hc] = f2bf(vh[row * 48 + hc]);
  }
  __syncthreads();
  // s_out = relu(concat(s,nrm) @ Ws^T + bs)
  gemm_grid(Ascat, w.kspad, 1, w.Ws, w.kspad, 8, w.kspad, wave, lane,
            [&](int row, int col, float val) {
              float x = val + w.bs[col];
              x = x > 0.f ? x : 0.f;
              if (sout_f) sout_f[row * 128 + col] = x;
              sout_b[row * ldSB + col] = f2bf(x);
            });
  __syncthreads();
  // gate = sigmoid(s_out @ Wg^T + bg)
  gemm_grid(sout_b, ldSB, 1, w.Wg, 128, 1, 128, wave, lane,
            [&](int row, int col, float val) {
              gate[row * 16 + col] = 1.f / (1.f + __expf(-(val + w.bg[col])));
            });
  __syncthreads();
  // vout = (Wu * vh) * gate
  gemm_grid(Avec, ldV, 3, w.Wu, w.hpad, 1, w.hpad, wave, lane,
            [&](int row, int col, float val) {
              int n = row & 15;
              vout[row * 16 + col] = val * gate[n * 16 + col];
            });
  __syncthreads();
}

// vfs = sqrt(sum_c (v2s_W @ v)^2 + eps); each wave owns a column tile, loops c
__device__ void v2s_norm(const unsigned short* Avec, int ldV, const unsigned short* Wv,
                         unsigned short* vfs_b, int wave, int lane) {
  const int half = (lane >> 4) & 1;
  const int l16 = lane & 15;
  for (int nt = wave; nt < 8; nt += WAVES) {
    float sq[8];
#pragma unroll
    for (int r = 0; r < 8; ++r) sq[r] = 0.f;
    for (int c = 0; c < 3; ++c) {
      v8f t = wmma_tile(Avec, ldV, c * 16, Wv, 32, nt * 16, 32, lane);
#pragma unroll
      for (int r = 0; r < 8; ++r) sq[r] += t[r] * t[r];
    }
#pragma unroll
    for (int r = 0; r < 8; ++r)
      vfs_b[(half * 8 + r) * 128 + nt * 16 + l16] = f2bf(sqrtf(sq[r] + 1e-8f));
  }
}

// ---------- LDS fill helpers ----------
__device__ void fill_vec_gathered(const float* xv, int V, const int* nodes,
                                  const float* rotL, unsigned short* A, int tid) {
  for (int idx = tid; idx < 48 * 64; idx += TPB) A[idx] = 0;
  __syncthreads();
  for (int idx = tid; idx < 16 * V; idx += TPB) {
    int n = idx / V, v = idx - n * V;
    const float* s = xv + (size_t)nodes[n] * (V * 3) + v * 3;
    const float* R = rotL + n * 9;
    float x = s[0], y = s[1], z = s[2];
    A[(0 * 16 + n) * 64 + v] = f2bf(x * R[0] + y * R[1] + z * R[2]);
    A[(1 * 16 + n) * 64 + v] = f2bf(x * R[3] + y * R[4] + z * R[5]);
    A[(2 * 16 + n) * 64 + v] = f2bf(x * R[6] + y * R[7] + z * R[8]);
  }
  __syncthreads();
}
__device__ void fill_vec_from_f32(const float* Vf, unsigned short* A, int tid) {
  for (int idx = tid; idx < 48 * 64; idx += TPB) A[idx] = 0;
  __syncthreads();
  for (int idx = tid; idx < 48 * 16; idx += TPB) {
    int row = idx >> 4, o = idx & 15;
    A[row * 64 + o] = f2bf(Vf[idx]);
  }
  __syncthreads();
}
__device__ void fill_scat_gathered(const float* xs, int cols, const int* nodes,
                                   unsigned short* A, int ldS, int tid) {
  for (int idx = tid; idx < 16 * ldS; idx += TPB) {
    int n = idx / ldS, c = idx - n * ldS;
    float v = (c < cols) ? xs[(size_t)nodes[n] * cols + c] : 0.f;
    A[idx] = f2bf(v);
  }
}

// ---------- small kernels ----------
__global__ void pack_bf16(const float* src, unsigned short* dst, int rows, int cols, int kpad) {
  int idx = blockIdx.x * blockDim.x + threadIdx.x;
  if (idx >= rows * kpad) return;
  int r = idx / kpad, k = idx - r * kpad;
  dst[idx] = f2bf(k < cols ? src[r * cols + k] : 0.f);
}
__global__ void init_kernel(float* out, float* m, float* denom, float* cnt, int N) {
  int idx = blockIdx.x * blockDim.x + threadIdx.x;
  if (idx < N * 176) out[idx] = 0.f;
  if (idx < N * 16) { m[idx] = __uint_as_float(0xFF800000u); denom[idx] = 0.f; }
  if (idx < N) cnt[idx] = 0.f;
}
__global__ void pass2_kernel(const int* ei, const float* m, float* scores,
                             float* denom, float* cnt, int E) {
  int idx = blockIdx.x * blockDim.x + threadIdx.x;
  if (idx >= E * 16) return;
  int e = idx >> 4, k = idx & 15;
  int i = ei[(size_t)E + e];
  float mm = m[(size_t)i * 16 + k];
  if (!(mm > -1e37f && mm < 1e37f)) mm = 0.f;
  float ex = __expf(scores[idx] - mm);
  scores[idx] = ex;
  atomicAdd(&denom[(size_t)i * 16 + k], ex);
  if (k == 0) atomicAdd(&cnt[i], 1.0f);
}
__global__ void finalize_kernel(float* out, const float* cnt, int N) {
  int idx = blockIdx.x * blockDim.x + threadIdx.x;
  if (idx >= N * 176) return;
  out[idx] /= fmaxf(cnt[idx / 176], 1.0f);
}

// ---------- pass 1: target/source SVPs -> 16 raw attention scores/edge + seg-max ----------
__global__ __launch_bounds__(TPB)
void pass1_kernel(const float* x_s, const float* x_v, const float* rot,
                  const int* ei, int E, WAll W, float* scores, float* m) {
  __shared__ int nodesJ[16], nodesI[16];
  __shared__ float rotL[16 * 9];
  __shared__ unsigned short Avec[48 * 64];
  __shared__ unsigned short Ascat[16 * 160];
  __shared__ float vh[3 * 16 * 48];
  __shared__ float gate[16 * 16];
  __shared__ float sfv[16 * 16];
  __shared__ unsigned short s_ti[16 * 128], s_sj[16 * 128], vfs[16 * 128];
  __shared__ float v_ti[3 * 16 * 16], v_sj[3 * 16 * 16];

  const int tid = threadIdx.x, wave = tid >> 5, lane = tid & 31;
  const int e0 = blockIdx.x * 16;
  if (tid < 16) {
    int e = e0 + tid; if (e >= E) e = E - 1;
    nodesJ[tid] = ei[e];
    nodesI[tid] = ei[(size_t)E + e];
  }
  __syncthreads();
  if (tid < 144) rotL[tid] = rot[(size_t)nodesI[tid / 9] * 9 + (tid % 9)];
  __syncthreads();

  // target SVP
  fill_scat_gathered(x_s, 128, nodesI, Ascat, 160, tid);
  fill_vec_gathered(x_v, 16, nodesI, rotL, Avec, tid);
  run_svp(W.tgt, Avec, 64, Ascat, vh, gate, nullptr, s_ti, 128, v_ti, wave, lane, tid);
  // source SVP
  fill_scat_gathered(x_s, 128, nodesJ, Ascat, 160, tid);
  fill_vec_gathered(x_v, 16, nodesJ, rotL, Avec, tid);
  run_svp(W.src, Avec, 64, Ascat, vh, gate, nullptr, s_sj, 128, v_sj, wave, lane, tid);
  // vfs, sfv
  fill_vec_from_f32(v_sj, Avec, tid);
  v2s_norm(Avec, 64, W.v2s, vfs, wave, lane);
  __syncthreads();
  gemm_grid(s_sj, 128, 1, W.s2v, 128, 1, 128, wave, lane,
            [&](int row, int col, float val) { sfv[row * 16 + col] = val; });
  __syncthreads();

  if (tid < 64) {
    int n = tid >> 2, h = tid & 3;
    float sc1 = 0.f, sc2 = 0.f;
    for (int d = 0; d < 32; ++d) {
      float t = bf2f(s_ti[n * 128 + h * 32 + d]);
      sc1 += t * bf2f(vfs[n * 128 + h * 32 + d]);
      sc2 += t * bf2f(s_sj[n * 128 + h * 32 + d]);
    }
    float sc3 = 0.f, sc4 = 0.f;
    for (int f = 0; f < 12; ++f) {
      int fl = h * 12 + f, o = fl / 3, c = fl - o * 3;
      float t = v_ti[(c * 16 + n) * 16 + o];
      sc3 += t * sfv[n * 16 + o];
      sc4 += t * v_sj[(c * 16 + n) * 16 + o];
    }
    int e = e0 + n;
    if (e < E) {
      size_t mb = (size_t)nodesI[n] * 16;
      size_t sb = (size_t)e * 16;
      scores[sb + 0 + h] = sc1;  atomicMaxF(&m[mb + 0 + h], sc1);
      scores[sb + 4 + h] = sc2;  atomicMaxF(&m[mb + 4 + h], sc2);
      scores[sb + 8 + h] = sc3;  atomicMaxF(&m[mb + 8 + h], sc3);
      scores[sb + 12 + h] = sc4; atomicMaxF(&m[mb + 12 + h], sc4);
    }
  }
}

// ---------- pass 3: attention-weighted message + out1/out2 SVPs + scatter ----------
__global__ __launch_bounds__(TPB)
void pass3_kernel(const float* x_s, const float* x_v, const float* edge_s,
                  const float* edge_v, const float* rot, const int* ei, int E,
                  WAll W, const float* scores, const float* denom, float* out) {
  __shared__ int nodesJ[16], nodesI[16];
  __shared__ float rotL[16 * 9];
  __shared__ unsigned short Avec[48 * 64];
  __shared__ unsigned short AscatE[16 * 64];
  __shared__ unsigned short Ascat1[16 * 448];
  __shared__ unsigned short Ascat2[16 * 160];
  __shared__ float vh[3 * 16 * 48];
  __shared__ float gate[16 * 16];
  __shared__ float sfv[16 * 16];
  __shared__ float attw[16 * 16];
  __shared__ float soutF[16 * 128];
  __shared__ unsigned short s_sj[16 * 128], vfs[16 * 128];
  __shared__ float v_sj[3 * 16 * 16], V0[3 * 16 * 16];

  const int tid = threadIdx.x, wave = tid >> 5, lane = tid & 31;
  const int e0 = blockIdx.x * 16;
  if (tid < 16) {
    int e = e0 + tid; if (e >= E) e = E - 1;
    nodesJ[tid] = ei[e];
    nodesI[tid] = ei[(size_t)E + e];
  }
  __syncthreads();
  if (tid < 144) rotL[tid] = rot[(size_t)nodesI[tid / 9] * 9 + (tid % 9)];
  // attention weights
  for (int idx = tid; idx < 256; idx += TPB) {
    int n = idx >> 4, k = idx & 15;
    int e = e0 + n; if (e >= E) e = E - 1;
    attw[idx] = scores[(size_t)e * 16 + k] /
                (denom[(size_t)nodesI[n] * 16 + k] + 1e-16f);
  }
  // zero pads of concat buffers
  for (int idx = tid; idx < 256; idx += TPB) {
    int n = idx >> 4, c = idx & 15;
    Ascat1[n * 448 + 432 + c] = 0;
    Ascat2[n * 160 + 144 + c] = 0;
  }
  __syncthreads();

  // edge SVP (first: s_e -> Ascat1[:,256:384], v_e2 -> V0)
  for (int idx = tid; idx < 16 * 64; idx += TPB) {
    int n = idx >> 6, c = idx & 63;
    int e = e0 + n; if (e >= E) e = E - 1;
    AscatE[idx] = f2bf(c < 32 ? edge_s[(size_t)e * 32 + c] : 0.f);
  }
  for (int idx = tid; idx < 48 * 64; idx += TPB) Avec[idx] = 0;
  __syncthreads();
  for (int idx = tid; idx < 16 * 4; idx += TPB) {
    int n = idx >> 2, v = idx & 3;
    int e = e0 + n; if (e >= E) e = E - 1;
    const float* s = edge_v + (size_t)e * 12 + v * 3;
    const float* R = rotL + n * 9;
    float x = s[0], y = s[1], z = s[2];
    Avec[(0 * 16 + n) * 64 + v] = f2bf(x * R[0] + y * R[1] + z * R[2]);
    Avec[(1 * 16 + n) * 64 + v] = f2bf(x * R[3] + y * R[4] + z * R[5]);
    Avec[(2 * 16 + n) * 64 + v] = f2bf(x * R[6] + y * R[7] + z * R[8]);
  }
  __syncthreads();
  run_svp(W.edge, Avec, 64, AscatE, vh, gate, nullptr, Ascat1 + 256, 448, V0, wave, lane, tid);

  // source SVP (recompute; scalar features staged in Ascat2 — stride 160 matches
  // src.kspad; Ascat2 is later overwritten by out1's s_out epilogue)
  fill_scat_gathered(x_s, 128, nodesJ, Ascat2, 160, tid);
  fill_vec_gathered(x_v, 16, nodesJ, rotL, Avec, tid);
  run_svp(W.src, Avec, 64, Ascat2, vh, gate, nullptr, s_sj, 128, v_sj, wave, lane, tid);

  // vfs / sfv
  fill_vec_from_f32(v_sj, Avec, tid);
  v2s_norm(Avec, 64, W.v2s, vfs, wave, lane);
  __syncthreads();
  gemm_grid(s_sj, 128, 1, W.s2v, 128, 1, 128, wave, lane,
            [&](int row, int col, float val) { sfv[row * 16 + col] = val; });
  __syncthreads();

  // attention-weighted concat: scalars -> Ascat1[:,0:256], vectors -> Avec[:,0:48]
  for (int idx = tid; idx < 16 * 256; idx += TPB) {
    int n = idx >> 8, c = idx & 255;
    if (c < 128) {
      float v = bf2f(vfs[n * 128 + c]) * attw[n * 16 + 0 + (c >> 5)];
      Ascat1[n * 448 + c] = f2bf(v);
    } else {
      int cc = c - 128;
      float v = bf2f(s_sj[n * 128 + cc]) * attw[n * 16 + 4 + (cc >> 5)];
      Ascat1[n * 448 + 128 + cc] = f2bf(v);
    }
  }
  for (int idx = tid; idx < 48 * 16; idx += TPB) {
    int row = idx >> 4, o = idx & 15;
    int c = row >> 4, n = row & 15;
    int hv = (o * 3 + c) / 12;
    Avec[row * 64 + o]      = f2bf(sfv[n * 16 + o] * attw[n * 16 + 8 + hv]);
    Avec[row * 64 + 16 + o] = f2bf(v_sj[row * 16 + o] * attw[n * 16 + 12 + hv]);
    Avec[row * 64 + 32 + o] = f2bf(V0[row * 16 + o]);  // v_e2 (no attention)
    Avec[row * 64 + 48 + o] = 0;
  }
  __syncthreads();

  // out1 SVP: s_out -> Ascat2[:,0:128] (input of out2), vout -> V0
  run_svp(W.out1, Avec, 64, Ascat1, vh, gate, nullptr, Ascat2, 160, V0, wave, lane, tid);
  // out2 SVP
  fill_vec_from_f32(V0, Avec, tid);
  run_svp(W.out2, Avec, 64, Ascat2, vh, gate, soutF, vfs, 128, v_sj, wave, lane, tid);

  // rotate back to global frame + scatter-add
  for (int idx = tid; idx < 16 * 16; idx += TPB) {
    int n = idx >> 4, o = idx & 15;
    int e = e0 + n;
    if (e >= E) continue;
    int tg = nodesI[n];
    float c0 = v_sj[(0 * 16 + n) * 16 + o];
    float c1 = v_sj[(1 * 16 + n) * 16 + o];
    float c2 = v_sj[(2 * 16 + n) * 16 + o];
    const float* R = rotL + n * 9;
    atomicAdd(&out[(size_t)tg * 176 + 128 + o * 3 + 0], c0 * R[0] + c1 * R[3] + c2 * R[6]);
    atomicAdd(&out[(size_t)tg * 176 + 128 + o * 3 + 1], c0 * R[1] + c1 * R[4] + c2 * R[7]);
    atomicAdd(&out[(size_t)tg * 176 + 128 + o * 3 + 2], c0 * R[2] + c1 * R[5] + c2 * R[8]);
  }
  for (int idx = tid; idx < 16 * 128; idx += TPB) {
    int n = idx >> 7, col = idx & 127;
    int e = e0 + n;
    if (e >= E) continue;
    atomicAdd(&out[(size_t)nodesI[n] * 176 + col], soutF[n * 128 + col]);
  }
}

// ---------- host launcher ----------
struct Job { int idx, rows, cols, kpad; };

extern "C" void kernel_launch(void* const* d_in, const int* in_sizes, int n_in,
                              void* d_out, int out_size, void* d_ws, size_t ws_size,
                              hipStream_t stream) {
  const float* x_s    = (const float*)d_in[0];
  const float* x_v    = (const float*)d_in[1];
  const float* edge_s = (const float*)d_in[2];
  const float* edge_v = (const float*)d_in[3];
  const float* rot    = (const float*)d_in[4];
  const int*   ei     = (const int*)d_in[37];
  const int N = in_sizes[0] / 128;
  const int E = in_sizes[37] / 2;
  float* out = (float*)d_out;

  static const Job jobs[22] = {
      {5, 16, 16, 32},   {6, 16, 16, 32},   {7, 128, 144, 160}, {9, 16, 128, 128},
      {11, 16, 16, 32},  {12, 16, 16, 32},  {13, 128, 144, 160},{15, 16, 128, 128},
      {17, 16, 4, 32},   {18, 16, 16, 32},  {19, 128, 48, 64},  {21, 16, 128, 128},
      {23, 48, 48, 64},  {24, 16, 48, 64},  {25, 128, 432, 448},{27, 16, 128, 128},
      {29, 16, 16, 32},  {30, 16, 16, 32},  {31, 128, 144, 160},{33, 16, 128, 128},
      {35, 128, 16, 32}, {36, 16, 128, 128}};

  unsigned short* wbase = (unsigned short*)d_ws;
  size_t offs[22], off = 0;
  for (int j = 0; j < 22; ++j) { offs[j] = off; off += (size_t)jobs[j].rows * jobs[j].kpad; }
  size_t wbytes = (off * 2 + 255) & ~(size_t)255;
  float* m      = (float*)((char*)d_ws + wbytes);
  float* denom  = m + (size_t)N * 16;
  float* cnt    = denom + (size_t)N * 16;
  float* scores = cnt + N;

  for (int j = 0; j < 22; ++j) {
    int tot = jobs[j].rows * jobs[j].kpad;
    pack_bf16<<<(tot + 255) / 256, 256, 0, stream>>>(
        (const float*)d_in[jobs[j].idx], wbase + offs[j], jobs[j].rows, jobs[j].cols,
        jobs[j].kpad);
  }
  init_kernel<<<(N * 176 + 255) / 256, 256, 0, stream>>>(out, m, denom, cnt, N);

  WAll W;
  W.tgt  = {wbase + offs[0],  wbase + offs[1],  wbase + offs[2],  wbase + offs[3],
            (const float*)d_in[8],  (const float*)d_in[10], 32, 16, 32, 128, 160};
  W.src  = {wbase + offs[4],  wbase + offs[5],  wbase + offs[6],  wbase + offs[7],
            (const float*)d_in[14], (const float*)d_in[16], 32, 16, 32, 128, 160};
  W.edge = {wbase + offs[8],  wbase + offs[9],  wbase + offs[10], wbase + offs[11],
            (const float*)d_in[20], (const float*)d_in[22], 32, 16, 32, 32, 64};
  W.out1 = {wbase + offs[12], wbase + offs[13], wbase + offs[14], wbase + offs[15],
            (const float*)d_in[26], (const float*)d_in[28], 64, 48, 64, 384, 448};
  W.out2 = {wbase + offs[16], wbase + offs[17], wbase + offs[18], wbase + offs[19],
            (const float*)d_in[32], (const float*)d_in[34], 32, 16, 32, 128, 160};
  W.v2s = wbase + offs[20];
  W.s2v = wbase + offs[21];

  const int ntiles = (E + 15) / 16;
  pass1_kernel<<<ntiles, TPB, 0, stream>>>(x_s, x_v, rot, ei, E, W, scores, m);
  pass2_kernel<<<(E * 16 + 255) / 256, 256, 0, stream>>>(ei, m, scores, denom, cnt, E);
  pass3_kernel<<<ntiles, TPB, 0, stream>>>(x_s, x_v, edge_s, edge_v, rot, ei, E, W,
                                           scores, denom, out);
  finalize_kernel<<<(N * 176 + 255) / 256, 256, 0, stream>>>(out, cnt, N);
}
